// MoEFSCIL_38912403701799
// MI455X (gfx1250) — matrix operand order repaired and checked by reading, not compiled, not run on hardware
//
#include <hip/hip_runtime.h>
#include <hip/hip_bf16.h>

#define DIMD 512
#define EEXP 4
#define HH 7
#define WW 7
#define LL 49
#define NHEADS 8
#define NSTATE 64
#define RANK 64
#define BB 8
#define KDIR 4
#define BL (BB * LL) /* 392 */

typedef __attribute__((ext_vector_type(16))) __bf16 v16bf;
typedef __attribute__((ext_vector_type(8)))  float  v8f;

__device__ __forceinline__ float siluf(float x) { return x / (1.f + __expf(-x)); }
__device__ __forceinline__ float softplusf(float x) { return x > 20.f ? x : log1pf(__expf(x)); }

// Build a 16-element bf16 fragment from 2x8 contiguous floats (A layout,
// contiguous K): K group 0 at kk + half*16, K group 1 at kk + half*16 + 8.
__device__ __forceinline__ v16bf load_afrag(const float* __restrict__ arow, int kk, int half)
{
    const float4* p0 = reinterpret_cast<const float4*>(arow + kk + half * 16);
    const float4* p1 = reinterpret_cast<const float4*>(arow + kk + half * 16 + 8);
    float4 a0 = p0[0], a1 = p0[1];
    float4 a2 = p1[0], a3 = p1[1];
    v16bf f;
    f[0] = (__bf16)a0.x;  f[1] = (__bf16)a0.y;  f[2] = (__bf16)a0.z;  f[3] = (__bf16)a0.w;
    f[4] = (__bf16)a1.x;  f[5] = (__bf16)a1.y;  f[6] = (__bf16)a1.z;  f[7] = (__bf16)a1.w;
    f[8] = (__bf16)a2.x;  f[9] = (__bf16)a2.y;  f[10] = (__bf16)a2.z; f[11] = (__bf16)a2.w;
    f[12] = (__bf16)a3.x; f[13] = (__bf16)a3.y; f[14] = (__bf16)a3.z; f[15] = (__bf16)a3.w;
    return f;
}

// B fragment for the weight case: 16 contiguous floats at Wrow + kk + half*16.
__device__ __forceinline__ v16bf load_bfrag_cont(const float* __restrict__ wrow, int kk, int half)
{
    const float4* q = reinterpret_cast<const float4*>(wrow + kk + half * 16);
    float4 w0 = q[0], w1 = q[1], w2 = q[2], w3 = q[3];
    v16bf f;
    f[0] = (__bf16)w0.x;  f[1] = (__bf16)w0.y;  f[2] = (__bf16)w0.z;  f[3] = (__bf16)w0.w;
    f[4] = (__bf16)w1.x;  f[5] = (__bf16)w1.y;  f[6] = (__bf16)w1.z;  f[7] = (__bf16)w1.w;
    f[8] = (__bf16)w2.x;  f[9] = (__bf16)w2.y;  f[10] = (__bf16)w2.z; f[11] = (__bf16)w2.w;
    f[12] = (__bf16)w3.x; f[13] = (__bf16)w3.y; f[14] = (__bf16)w3.z; f[15] = (__bf16)w3.w;
    return f;
}

// ---------------------------------------------------------------------------
// Wave-tiled GEMM on the WMMA pipe. Each wave owns a 32x32 output macro-tile:
// 2 A-fragments x 2 B-fragments -> 4 v_wmma_f32_16x16x32_bf16 per K-step.
//   C[M,N] = act( A x B + bias )
//   A(m,k) = A[m*lda + k]                       (always contiguous in k)
//   BKCONT=true : B(k,n) = Bp[n*ldb + k]        (weight, contiguous in k)
//   BKCONT=false: B(k,n) = Bp[k*ldb + n]        (activation, contiguous in n)
// biasAxis: 0 none, 1 over N, 2 over M. act: 0 none, 1 softplus.
// Out-of-range rows/cols are clamped for loads (values land in masked-out
// outputs), so all fragment loads are unconditional and vectorizable.
// ---------------------------------------------------------------------------
template <bool BKCONT>
__global__ void moe_wgemm_t(const float* __restrict__ A, int lda,
                            const float* __restrict__ Bp, int ldb,
                            float* __restrict__ C, int ldc,
                            const float* __restrict__ bias, int biasAxis, int act,
                            int M, int N, int K)
{
    const int tilesN = (N + 31) >> 5;
    const int tilesM = (M + 31) >> 5;
    int tile = blockIdx.x * blockDim.y + threadIdx.y;   // uniform per wave
    if (tile >= tilesM * tilesN) return;
    int tm = tile / tilesN, tn = tile - tm * tilesN;
    int lane = threadIdx.x & 31;
    int half = lane >> 4, li = lane & 15;

    int row0 = tm * 32 + li, row1 = row0 + 16;
    int col0 = tn * 32 + li, col1 = col0 + 16;
    int rc0 = (row0 < M) ? row0 : 0;
    int rc1 = (row1 < M) ? row1 : 0;
    int cc0 = (col0 < N) ? col0 : 0;
    int cc1 = (col1 < N) ? col1 : 0;
    const float* arow0 = A + (size_t)rc0 * lda;
    const float* arow1 = A + (size_t)rc1 * lda;

    v8f acc00 = {}, acc01 = {}, acc10 = {}, acc11 = {};

    for (int kk = 0; kk < K; kk += 32) {
        if (kk + 32 < K) {
            __builtin_prefetch(arow0 + kk + 32, 0, 3);
            __builtin_prefetch(arow1 + kk + 32, 0, 3);
            if (BKCONT) {
                __builtin_prefetch(Bp + (size_t)cc0 * ldb + kk + 32, 0, 3);
                __builtin_prefetch(Bp + (size_t)cc1 * ldb + kk + 32, 0, 3);
            } else {
                __builtin_prefetch(Bp + (size_t)(kk + 32) * ldb + cc0, 0, 3);
            }
        }
        v16bf a0 = load_afrag(arow0, kk, half);
        v16bf a1 = load_afrag(arow1, kk, half);

        // B 32x16 bf16 fragment, contiguous K order as for A: lane half holds K=half*16..half*16+15
        v16bf b0, b1;
        if (BKCONT) {
            b0 = load_bfrag_cont(Bp + (size_t)cc0 * ldb, kk, half);
            b1 = load_bfrag_cont(Bp + (size_t)cc1 * ldb, kk, half);
        } else {
            float t0[16], t1[16];
            #pragma unroll
            for (int j = 0; j < 16; ++j) {
                int k = kk + half * 16 + j;
                t0[j] = Bp[(size_t)k * ldb + cc0];
                t1[j] = Bp[(size_t)k * ldb + cc1];
            }
            #pragma unroll
            for (int j = 0; j < 16; ++j) { b0[j] = (__bf16)t0[j]; b1[j] = (__bf16)t1[j]; }
        }

        acc00 = __builtin_amdgcn_wmma_f32_16x16x32_bf16(false, a0, false, b0, (short)0, acc00, false, false);
        acc01 = __builtin_amdgcn_wmma_f32_16x16x32_bf16(false, a0, false, b1, (short)0, acc01, false, false);
        acc10 = __builtin_amdgcn_wmma_f32_16x16x32_bf16(false, a1, false, b0, (short)0, acc10, false, false);
        acc11 = __builtin_amdgcn_wmma_f32_16x16x32_bf16(false, a1, false, b1, (short)0, acc11, false, false);
    }

    // D layout: VGPR r -> M = r + half*8, N = li
    #pragma unroll
    for (int rt = 0; rt < 2; ++rt) {
        #pragma unroll
        for (int ct = 0; ct < 2; ++ct) {
            const v8f& acc = rt ? (ct ? acc11 : acc10) : (ct ? acc01 : acc00);
            int n = tn * 32 + ct * 16 + li;
            #pragma unroll
            for (int r = 0; r < 8; ++r) {
                int m = tm * 32 + rt * 16 + half * 8 + r;
                if (m < M && n < N) {
                    float v = acc[r];
                    if (biasAxis == 1) v += bias[n];
                    else if (biasAxis == 2) v += bias[m];
                    if (act == 1) v = softplusf(v);
                    C[(size_t)m * ldc + n] = v;
                }
            }
        }
    }
}

// ---------------------------------------------------------------------------
// Self-attention: one block per (b,h); thread = query index.
// ---------------------------------------------------------------------------
__global__ void moe_attn(const float* __restrict__ qkv, float* __restrict__ o)
{
    int b = blockIdx.x / NHEADS, h = blockIdx.x % NHEADS;
    int q = threadIdx.x;
    if (q >= LL) return;
    const float* qp = qkv + (b * LL + q) * (3 * DIMD) + h * 64;
    float s[LL];
    float mx = -1e30f;
    for (int k = 0; k < LL; ++k) {
        const float* kp = qkv + (b * LL + k) * (3 * DIMD) + DIMD + h * 64;
        float d = 0.f;
        #pragma unroll 8
        for (int j = 0; j < 64; ++j) d += qp[j] * kp[j];
        d *= 0.125f;
        s[k] = d;
        mx = fmaxf(mx, d);
    }
    float ssum = 0.f;
    for (int k = 0; k < LL; ++k) { s[k] = __expf(s[k] - mx); ssum += s[k]; }
    float inv = 1.f / ssum;
    for (int j = 0; j < 64; ++j) {
        float a = 0.f;
        for (int k = 0; k < LL; ++k)
            a += s[k] * qkv[(b * LL + k) * (3 * DIMD) + 2 * DIMD + h * 64 + j];
        o[(b * LL + q) * DIMD + h * 64 + j] = a * inv;
    }
}

// ---------------------------------------------------------------------------
// Cross-attention weights: per token, softmax over E per head, mean over heads.
// ---------------------------------------------------------------------------
__global__ void moe_crossaw(const float* __restrict__ q2, const float* __restrict__ k2,
                            float* __restrict__ aw)
{
    int t = blockIdx.x * blockDim.x + threadIdx.x;
    if (t >= BL) return;
    float acc[EEXP] = {0.f, 0.f, 0.f, 0.f};
    for (int h = 0; h < NHEADS; ++h) {
        float sc[EEXP], mx = -1e30f;
        #pragma unroll
        for (int e = 0; e < EEXP; ++e) {
            float d = 0.f;
            for (int j = 0; j < 64; ++j)
                d += q2[t * DIMD + h * 64 + j] * k2[e * DIMD + h * 64 + j];
            sc[e] = d * 0.125f;
            mx = fmaxf(mx, sc[e]);
        }
        float ss = 0.f;
        #pragma unroll
        for (int e = 0; e < EEXP; ++e) { sc[e] = __expf(sc[e] - mx); ss += sc[e]; }
        #pragma unroll
        for (int e = 0; e < EEXP; ++e) acc[e] += sc[e] / ss;
    }
    #pragma unroll
    for (int e = 0; e < EEXP; ++e) aw[t * EEXP + e] = acc[e] * (1.f / NHEADS);
}

// ---------------------------------------------------------------------------
// Gate: raw softmax, top-2 mask, capacity normalize, aux loss, dense weights.
// ---------------------------------------------------------------------------
__global__ void moe_gate(const float* __restrict__ aw, float* __restrict__ dense,
                         float* __restrict__ auxOut)
{
    if (threadIdx.x != 0 || blockIdx.x != 0) return;
    float raw[BB][EEXP], mask[BB][EEXP], gs[BB][EEXP];
    for (int b = 0; b < BB; ++b) {
        float m[EEXP], mx = -1e30f;
        for (int e = 0; e < EEXP; ++e) {
            float s = 0.f;
            for (int l = 0; l < LL; ++l) s += aw[(b * LL + l) * EEXP + e];
            m[e] = s / LL;
            mx = fmaxf(mx, m[e]);
        }
        float ss = 0.f;
        for (int e = 0; e < EEXP; ++e) { m[e] = __expf(m[e] - mx); ss += m[e]; }
        for (int e = 0; e < EEXP; ++e) raw[b][e] = m[e] / ss;
    }
    for (int b = 0; b < BB; ++b) {
        int i0 = 0;
        for (int e = 1; e < EEXP; ++e) if (raw[b][e] > raw[b][i0]) i0 = e;
        int i1 = (i0 == 0) ? 1 : 0;
        for (int e = 0; e < EEXP; ++e) if (e != i0 && raw[b][e] > raw[b][i1]) i1 = e;
        for (int e = 0; e < EEXP; ++e) mask[b][e] = (e == i0 || e == i1) ? 1.f : 0.f;
    }
    const float cap = 10.f;  // int(1.25 * 8)
    for (int e = 0; e < EEXP; ++e) {
        float denom = 1e-6f;
        for (int b = 0; b < BB; ++b) denom += raw[b][e] * mask[b][e];
        for (int b = 0; b < BB; ++b) gs[b][e] = raw[b][e] * mask[b][e] / denom * cap;
    }
    float aux = 0.f;
    for (int e = 0; e < EEXP; ++e) {
        float rm = 0.f, mm = 0.f;
        for (int b = 0; b < BB; ++b) { rm += raw[b][e]; mm += mask[b][e]; }
        aux += (rm / BB) * (mm / BB);
    }
    *auxOut = 0.01f * (aux / EEXP) * (float)(EEXP * EEXP);
    for (int b = 0; b < BB; ++b) {
        int i0 = 0;
        for (int e = 1; e < EEXP; ++e) if (gs[b][e] > gs[b][i0]) i0 = e;
        int i1 = (i0 == 0) ? 1 : 0;
        for (int e = 0; e < EEXP; ++e) if (e != i0 && gs[b][e] > gs[b][i1]) i1 = e;
        float v0 = gs[b][i0], v1 = gs[b][i1];
        float mx = fmaxf(v0, v1);
        float e0 = __expf(v0 - mx), e1 = __expf(v1 - mx);
        float w0 = e0 / (e0 + e1), w1 = e1 / (e0 + e1);
        for (int e = 0; e < EEXP; ++e) dense[b * EEXP + e] = 0.f;
        dense[b * EEXP + i0] += w0;
        dense[b * EEXP + i1] += w1;
    }
}

// ---------------------------------------------------------------------------
// Depthwise 3x3 conv (SAME) + bias + SiLU. Output xconv layout: (b, c, l).
// ---------------------------------------------------------------------------
__global__ void moe_conv(const float* __restrict__ xz, const float* __restrict__ conv_w,
                         const float* __restrict__ conv_b, float* __restrict__ xconv, int e)
{
    int tid = blockIdx.x * blockDim.x + threadIdx.x;
    if (tid >= BB * DIMD * LL) return;
    int l = tid % LL;
    int c = (tid / LL) % DIMD;
    int b = tid / (LL * DIMD);
    int y = l / WW, x = l % WW;
    const float* w = conv_w + (e * DIMD + c) * 9;
    float acc = 0.f;
    #pragma unroll
    for (int ky = 0; ky < 3; ++ky) {
        int yy = y + ky - 1;
        if (yy < 0 || yy >= HH) continue;
        #pragma unroll
        for (int kx = 0; kx < 3; ++kx) {
            int xx = x + kx - 1;
            if (xx < 0 || xx >= WW) continue;
            acc += xz[(b * LL + yy * WW + xx) * (2 * DIMD) + c] * w[ky * 3 + kx];
        }
    }
    acc += conv_b[e * DIMD + c];
    xconv[(b * DIMD + c) * LL + l] = siluf(acc);
}

// ---------------------------------------------------------------------------
// 4-direction scan permutations: xs[(kd,c,b*49+l)] = xconv[b,c,perm_kd(l)].
// ---------------------------------------------------------------------------
__global__ void moe_dirs(const float* __restrict__ xconv, float* __restrict__ xs)
{
    int tid = blockIdx.x * blockDim.x + threadIdx.x;
    if (tid >= KDIR * DIMD * BL) return;
    int n = tid % BL;
    int c = (tid / BL) % DIMD;
    int kd = tid / (BL * DIMD);
    int b = n / LL, l = n % LL;
    int pl;
    if (kd == 0)      pl = l;
    else if (kd == 1) pl = 48 - l;
    else if (kd == 2) pl = (l % 7) * 7 + l / 7;
    else { int i2 = 48 - l; pl = (i2 % 7) * 7 + i2 / 7; }
    xs[tid] = xconv[(b * DIMD + c) * LL + pl];
}

// ---------------------------------------------------------------------------
// Selective scan: one thread per (kd, c, b); 64 states in registers.
// ---------------------------------------------------------------------------
__global__ void moe_scan(const float* __restrict__ delta, const float* __restrict__ xs,
                         const float* __restrict__ dbl, const float* __restrict__ A_log,
                         const float* __restrict__ Ds, float* __restrict__ ys, int e)
{
    int tid = blockIdx.x * blockDim.x + threadIdx.x;
    if (tid >= KDIR * DIMD * BB) return;
    int b = tid % BB;
    int c = (tid / BB) % DIMD;
    int kd = tid / (BB * DIMD);
    const float* dl = delta + (kd * DIMD + c) * BL + b * LL;
    const float* ul = xs + (kd * DIMD + c) * BL + b * LL;
    const float* Bb = dbl + kd * 192 * BL + RANK * BL + b * LL;
    const float* Cb = Bb + NSTATE * BL;
    const float* al = A_log + ((e * KDIR + kd) * DIMD + c) * NSTATE;
    float Aa[NSTATE], h[NSTATE];
    #pragma unroll
    for (int n = 0; n < NSTATE; ++n) { Aa[n] = -__expf(al[n]); h[n] = 0.f; }
    float Dv = Ds[(e * KDIR + kd) * DIMD + c];
    float* yo = ys + (kd * DIMD + c) * BL + b * LL;
    for (int t = 0; t < LL; ++t) {
        float dt = dl[t], u = ul[t];
        float du = dt * u;
        float y = 0.f;
        #pragma unroll
        for (int n = 0; n < NSTATE; ++n) {
            h[n] = __expf(dt * Aa[n]) * h[n] + du * Bb[n * BL + t];
            y += h[n] * Cb[n * BL + t];
        }
        yo[t] = y + Dv * u;
    }
}

// ---------------------------------------------------------------------------
// Inverse-permute + sum over directions, LayerNorm over channels (LDS tree
// reduction), multiply by SiLU(z). One block per (b,l), 256 threads x 2 ch.
// ---------------------------------------------------------------------------
__global__ void moe_lnmul(const float* __restrict__ ys, const float* __restrict__ xz,
                          const float* __restrict__ on_g, const float* __restrict__ on_b,
                          float* __restrict__ yact, int e)
{
    int bl = blockIdx.x;
    int b = bl / LL, l = bl % LL;
    int t2 = (l % 7) * 7 + l / 7;
    int idx[KDIR] = { l, 48 - l, t2, 48 - t2 };
    __shared__ float red[256];
    float yv[2];
    #pragma unroll
    for (int i = 0; i < 2; ++i) {
        int c = threadIdx.x + i * 256;
        float s = 0.f;
        #pragma unroll
        for (int kd = 0; kd < KDIR; ++kd)
            s += ys[kd * DIMD * BL + c * BL + b * LL + idx[kd]];
        yv[i] = s;
    }
    red[threadIdx.x] = yv[0] + yv[1];
    __syncthreads();
    for (int o = 128; o > 0; o >>= 1) {
        if (threadIdx.x < o) red[threadIdx.x] += red[threadIdx.x + o];
        __syncthreads();
    }
    float mean = red[0] * (1.f / DIMD);
    __syncthreads();
    float d0 = yv[0] - mean, d1 = yv[1] - mean;
    red[threadIdx.x] = d0 * d0 + d1 * d1;
    __syncthreads();
    for (int o = 128; o > 0; o >>= 1) {
        if (threadIdx.x < o) red[threadIdx.x] += red[threadIdx.x + o];
        __syncthreads();
    }
    float inv = rsqrtf(red[0] * (1.f / DIMD) + 1e-5f);
    #pragma unroll
    for (int i = 0; i < 2; ++i) {
        int c = threadIdx.x + i * 256;
        float v = (yv[i] - mean) * inv * on_g[e * DIMD + c] + on_b[e * DIMD + c];
        float z = xz[(b * LL + l) * (2 * DIMD) + DIMD + c];
        yact[(b * LL + l) * DIMD + c] = v * siluf(z);
    }
}

__global__ void moe_pool(const float* __restrict__ yact, float* __restrict__ pooled)
{
    int tid = blockIdx.x * blockDim.x + threadIdx.x;
    if (tid >= BB * DIMD) return;
    int b = tid / DIMD, c = tid % DIMD;
    float s = 0.f;
    for (int l = 0; l < LL; ++l) s += yact[(b * LL + l) * DIMD + c];
    pooled[tid] = s * (1.f / LL);
}

__global__ void moe_lnpool(const float* __restrict__ pooled, const float* __restrict__ ln_g,
                           const float* __restrict__ ln_b, float* __restrict__ outs, int e)
{
    int b = blockIdx.x;
    __shared__ float red[256];
    float v[2];
    #pragma unroll
    for (int i = 0; i < 2; ++i) v[i] = pooled[b * DIMD + threadIdx.x + i * 256];
    red[threadIdx.x] = v[0] + v[1];
    __syncthreads();
    for (int o = 128; o > 0; o >>= 1) {
        if (threadIdx.x < o) red[threadIdx.x] += red[threadIdx.x + o];
        __syncthreads();
    }
    float mean = red[0] * (1.f / DIMD);
    __syncthreads();
    float d0 = v[0] - mean, d1 = v[1] - mean;
    red[threadIdx.x] = d0 * d0 + d1 * d1;
    __syncthreads();
    for (int o = 128; o > 0; o >>= 1) {
        if (threadIdx.x < o) red[threadIdx.x] += red[threadIdx.x + o];
        __syncthreads();
    }
    float inv = rsqrtf(red[0] * (1.f / DIMD) + 1e-5f);
    #pragma unroll
    for (int i = 0; i < 2; ++i) {
        int c = threadIdx.x + i * 256;
        outs[(e * BB + b) * DIMD + c] =
            (v[i] - mean) * inv * ln_g[e * DIMD + c] + ln_b[e * DIMD + c];
    }
}

__global__ void moe_mix(const float* __restrict__ dense, const float* __restrict__ outs,
                        float* __restrict__ out)
{
    int tid = blockIdx.x * blockDim.x + threadIdx.x;
    if (tid >= BB * DIMD) return;
    int b = tid / DIMD, c = tid % DIMD;
    float s = 0.f;
    #pragma unroll
    for (int e = 0; e < EEXP; ++e)
        s += dense[b * EEXP + e] * outs[(e * BB + b) * DIMD + c];
    out[tid] = s;
}

// ---------------------------------------------------------------------------
extern "C" void kernel_launch(void* const* d_in, const int* in_sizes, int n_in,
                              void* d_out, int out_size, void* d_ws, size_t ws_size,
                              hipStream_t stream)
{
    (void)in_sizes; (void)n_in; (void)out_size; (void)ws_size;
    const float* x        = (const float*)d_in[0];
    const float* sa_in_w  = (const float*)d_in[1];
    const float* sa_in_b  = (const float*)d_in[2];
    const float* sa_out_w = (const float*)d_in[3];
    const float* sa_out_b = (const float*)d_in[4];
    const float* ca_in_w  = (const float*)d_in[5];
    const float* ca_in_b  = (const float*)d_in[6];
    const float* eq       = (const float*)d_in[7];
    const float* ein_w    = (const float*)d_in[8];
    const float* ein_b    = (const float*)d_in[9];
    const float* conv_w   = (const float*)d_in[10];
    const float* conv_b   = (const float*)d_in[11];
    const float* xproj_w  = (const float*)d_in[12];
    const float* dtproj_w = (const float*)d_in[13];
    const float* dtproj_b = (const float*)d_in[14];
    const float* A_log    = (const float*)d_in[15];
    const float* Ds       = (const float*)d_in[16];
    const float* on_g     = (const float*)d_in[17];
    const float* on_b     = (const float*)d_in[18];
    const float* ln_g     = (const float*)d_in[19];
    const float* ln_b     = (const float*)d_in[20];

    // workspace carve-up (floats); total ~4.75M floats (~19 MB)
    float* ws     = (float*)d_ws;
    float* qkv    = ws;                         // 392*1536
    float* obuf   = qkv + BL * 3 * DIMD;        // 392*512
    float* ctx    = obuf + BL * DIMD;           // 392*512
    float* q2     = ctx + BL * DIMD;            // 392*512
    float* k2     = q2 + BL * DIMD;             // 4*512
    float* aw     = k2 + EEXP * DIMD;           // 392*4
    float* dense  = aw + BL * EEXP;             // 32 (+pad)
    float* outs   = dense + 64;                 // 4*8*512
    float* xzb    = outs + EEXP * BB * DIMD;    // 392*1024  (per-expert, reused)
    float* xconv  = xzb + BL * 2 * DIMD;        // 8*512*49
    float* xsdir  = xconv + BB * DIMD * LL;     // 4*512*392
    float* dblb   = xsdir + KDIR * DIMD * BL;   // 4*192*392
    float* deltab = dblb + KDIR * 192 * BL;     // 4*512*392
    float* ysb    = deltab + KDIR * DIMD * BL;  // 4*512*392
    float* yact   = ysb + KDIR * DIMD * BL;     // 392*512
    float* pooled = yact + BL * DIMD;           // 8*512

    dim3 gb(32, 4);   // 4 waves per block, one 32x32 macro-tile per wave
    auto tgrid = [](int M, int N) {
        return (((M + 31) / 32) * ((N + 31) / 32) + 3) / 4;
    };

    // ---- gate path ----
    moe_wgemm_t<true><<<tgrid(BL, 3 * DIMD), gb, 0, stream>>>(
        x, DIMD, sa_in_w, DIMD, qkv, 3 * DIMD, sa_in_b, 1, 0, BL, 3 * DIMD, DIMD);
    moe_attn<<<BB * NHEADS, 64, 0, stream>>>(qkv, obuf);
    moe_wgemm_t<true><<<tgrid(BL, DIMD), gb, 0, stream>>>(
        obuf, DIMD, sa_out_w, DIMD, ctx, DIMD, sa_out_b, 1, 0, BL, DIMD, DIMD);
    moe_wgemm_t<true><<<tgrid(BL, DIMD), gb, 0, stream>>>(
        ctx, DIMD, ca_in_w, DIMD, q2, DIMD, ca_in_b, 1, 0, BL, DIMD, DIMD);
    moe_wgemm_t<true><<<tgrid(EEXP, DIMD), gb, 0, stream>>>(
        eq, DIMD, ca_in_w + DIMD * DIMD, DIMD, k2, DIMD, ca_in_b + DIMD, 1, 0,
        EEXP, DIMD, DIMD);
    moe_crossaw<<<(BL + 127) / 128, 128, 0, stream>>>(q2, k2, aw);
    moe_gate<<<1, 32, 0, stream>>>(aw, dense, ((float*)d_out) + BB * DIMD);

    // ---- experts (sequential, workspace reused) ----
    for (int e = 0; e < EEXP; ++e) {
        moe_wgemm_t<true><<<tgrid(BL, 2 * DIMD), gb, 0, stream>>>(
            x, DIMD, ein_w + (size_t)e * 2 * DIMD * DIMD, DIMD,
            xzb, 2 * DIMD, ein_b + e * 2 * DIMD, 1, 0, BL, 2 * DIMD, DIMD);
        moe_conv<<<(BB * DIMD * LL + 255) / 256, 256, 0, stream>>>(
            xzb, conv_w, conv_b, xconv, e);
        moe_dirs<<<(KDIR * DIMD * BL + 255) / 256, 256, 0, stream>>>(xconv, xsdir);
        for (int kd = 0; kd < KDIR; ++kd) {
            // dbl[kd] = xproj_w[e,kd] (192x512) @ xs[kd] (512 x 392)
            moe_wgemm_t<false><<<tgrid(192, BL), gb, 0, stream>>>(
                xproj_w + (size_t)(e * KDIR + kd) * 192 * DIMD, DIMD,
                xsdir + (size_t)kd * DIMD * BL, BL,
                dblb + (size_t)kd * 192 * BL, BL, nullptr, 0, 0, 192, BL, DIMD);
            // delta[kd] = softplus(dtproj_w[e,kd] (512x64) @ dt (64x392) + b_d)
            moe_wgemm_t<false><<<tgrid(DIMD, BL), gb, 0, stream>>>(
                dtproj_w + (size_t)(e * KDIR + kd) * DIMD * RANK, RANK,
                dblb + (size_t)kd * 192 * BL, BL,
                deltab + (size_t)kd * DIMD * BL, BL,
                dtproj_b + (e * KDIR + kd) * DIMD, 2, 1, DIMD, BL, RANK);
        }
        moe_scan<<<(KDIR * DIMD * BB + 255) / 256, 256, 0, stream>>>(
            deltab, xsdir, dblb, A_log, Ds, ysb, e);
        moe_lnmul<<<BL, 256, 0, stream>>>(ysb, xzb, on_g, on_b, yact, e);
        moe_pool<<<(BB * DIMD + 255) / 256, 256, 0, stream>>>(yact, pooled);
        moe_lnpool<<<BB, 256, 0, stream>>>(pooled, ln_g, ln_b, outs, e);
    }

    // ---- final mix ----
    moe_mix<<<(BB * DIMD + 255) / 256, 256, 0, stream>>>(dense, outs, (float*)d_out);
}
